// CausalSelfAttention_63058709840004
// MI455X (gfx1250) — compile-verified, matrix operand order checked by reading, not run on hardware
//
#include <hip/hip_runtime.h>
#include <hip/hip_bf16.h>

#define T_SEQ 4096
#define DMODEL 768
#define NH 6
#define HD 128
#define ATTN_SCALE 0.12f
#define RMS_EPS 1.1920929e-07f

typedef __bf16 v16bf __attribute__((ext_vector_type(16)));
typedef __bf16 v8bf  __attribute__((ext_vector_type(8)));
typedef float  v8f   __attribute__((ext_vector_type(8)));

// Load one 16x32 bf16 A-fragment (or 32x16 B-fragment via its transpose view).
// Per CDNA5 ISA: lane l holds row r (passed in), K elements
// { kg*8 .. kg*8+7 } U { 16+kg*8 .. 16+kg*8+7 }, kg = l>>4.
// => two contiguous 16-byte loads per lane.
static __device__ __forceinline__ v16bf frag_ld(const __bf16* base, int r, int ld,
                                                int kbase, int kg) {
  const v8bf lo = *(const v8bf*)(base + (size_t)r * ld + kbase + kg * 8);
  const v8bf hi = *(const v8bf*)(base + (size_t)r * ld + kbase + 16 + kg * 8);
  return __builtin_shufflevector(lo, hi, 0, 1, 2, 3, 4, 5, 6, 7,
                                 8, 9, 10, 11, 12, 13, 14, 15);
}

static __device__ __forceinline__ v8f wmma_bf16(v16bf a, v16bf b, v8f c) {
  return __builtin_amdgcn_wmma_f32_16x16x32_bf16(false, a, false, b,
                                                 (short)0, c, false, false);
}

// ---------------------------------------------------------------------------
// Kernels 1 & 4: C[M,N] = A[M,K] (f32) @ W[N,K]^T (f32), bf16 WMMA, f32 accum.
// Compile-time N,K so all addressing folds to base + immediate offsets.
// Block: 256 threads = 8 waves; block tile 128(M) x 64(N); wave tile 32x32;
// K-tile 64 (two WMMA k-steps per LDS stage).
// ---------------------------------------------------------------------------
template <int N, int K>
__global__ __launch_bounds__(256) void gemm_bf16_wmma(const float* __restrict__ A,
                                                      const float* __restrict__ W,
                                                      float* __restrict__ C) {
  __shared__ __bf16 As[128 * 64];  // 16 KB
  __shared__ __bf16 Bs[64 * 64];   // 8 KB

  const int tid  = threadIdx.x;
  const int lane = tid & 31;
  const int w    = tid >> 5;
  const int wm   = w >> 1;       // 0..3
  const int wn   = w & 1;        // 0..1
  const int bm   = blockIdx.y * 128;
  const int bn   = blockIdx.x * 64;
  const int nl   = lane & 15;
  const int kg   = lane >> 4;

  v8f acc[2][2];
#pragma unroll
  for (int i = 0; i < 2; ++i)
#pragma unroll
    for (int j = 0; j < 2; ++j) acc[i][j] = {};

  const int colA = (tid & 15) * 4;  // 0,4,..,60
  const int rowA = tid >> 4;        // 0..15

  for (int kb = 0; kb < K; kb += 64) {
    // stage A tile: 128 x 64 (f32 -> bf16), 8 x float4 per thread
#pragma unroll
    for (int r = 0; r < 8; ++r) {
      const int row = rowA + r * 16;
      const float4 f = *(const float4*)(A + (size_t)(bm + row) * K + kb + colA);
      As[row * 64 + colA + 0] = (__bf16)f.x;
      As[row * 64 + colA + 1] = (__bf16)f.y;
      As[row * 64 + colA + 2] = (__bf16)f.z;
      As[row * 64 + colA + 3] = (__bf16)f.w;
    }
    // stage B^T tile (rows of W): 64 x 64, 4 x float4 per thread
#pragma unroll
    for (int r = 0; r < 4; ++r) {
      const int row = rowA + r * 16;
      const float4 f = *(const float4*)(W + (size_t)(bn + row) * K + kb + colA);
      Bs[row * 64 + colA + 0] = (__bf16)f.x;
      Bs[row * 64 + colA + 1] = (__bf16)f.y;
      Bs[row * 64 + colA + 2] = (__bf16)f.z;
      Bs[row * 64 + colA + 3] = (__bf16)f.w;
    }
    // keep HBM/L2 ahead of the matrix pipes: prefetch next K-tile
    if (kb + 64 < K) {
      __builtin_prefetch(A + (size_t)(bm + rowA) * K + kb + 64 + colA, 0, 0);
      __builtin_prefetch(W + (size_t)(bn + rowA) * K + kb + 64 + colA, 0, 0);
    }
    __syncthreads();

#pragma unroll
    for (int ks = 0; ks < 2; ++ks) {
      const v16bf a0 = frag_ld(As, wm * 32 + nl, 64, ks * 32, kg);
      const v16bf a1 = frag_ld(As, wm * 32 + 16 + nl, 64, ks * 32, kg);
      const v16bf b0 = frag_ld(Bs, wn * 32 + nl, 64, ks * 32, kg);
      const v16bf b1 = frag_ld(Bs, wn * 32 + 16 + nl, 64, ks * 32, kg);
      acc[0][0] = wmma_bf16(a0, b0, acc[0][0]);
      acc[0][1] = wmma_bf16(a0, b1, acc[0][1]);
      acc[1][0] = wmma_bf16(a1, b0, acc[1][0]);
      acc[1][1] = wmma_bf16(a1, b1, acc[1][1]);
    }
    __syncthreads();
  }

  // one base pointer, all else immediate offsets (N,K compile-time)
  float* cbase = C + (size_t)(bm + wm * 32 + 8 * kg) * N + bn + wn * 32 + nl;
#pragma unroll
  for (int mt = 0; mt < 2; ++mt)
#pragma unroll
    for (int nt = 0; nt < 2; ++nt)
#pragma unroll
      for (int j = 0; j < 8; ++j)
        cbase[(size_t)(mt * 16 + j) * N + nt * 16] = acc[mt][nt][j];
}

// ---------------------------------------------------------------------------
// Kernel 2: per (t,h) fused RMSNorm + rotary + V mix; emit bf16 head-major.
// ---------------------------------------------------------------------------
__global__ __launch_bounds__(128) void qkv_epilogue(const float* __restrict__ qkv,
                                                    const float* __restrict__ ve,
                                                    const float* __restrict__ lambdas,
                                                    __bf16* __restrict__ Qb,
                                                    __bf16* __restrict__ Kb,
                                                    __bf16* __restrict__ Vb) {
  __shared__ float sredq[4], sredk[4];
  __shared__ float sq[128], sk[128];

  const int t = blockIdx.x;
  const int h = blockIdx.y;
  const int d = threadIdx.x;

  const float* base = qkv + (size_t)t * (3 * DMODEL);
  const float q = base[h * HD + d];
  const float k = base[DMODEL + h * HD + d];
  const float v = base[2 * DMODEL + h * HD + d];

  float qs = q * q, ks = k * k;
#pragma unroll
  for (int off = 1; off < 32; off <<= 1) {
    qs += __shfl_xor(qs, off);
    ks += __shfl_xor(ks, off);
  }
  if ((d & 31) == 0) { sredq[d >> 5] = qs; sredk[d >> 5] = ks; }
  __syncthreads();
  const float qss = sredq[0] + sredq[1] + sredq[2] + sredq[3];
  const float kss = sredk[0] + sredk[1] + sredk[2] + sredk[3];
  const float qn = q * rsqrtf(qss * (1.0f / HD) + RMS_EPS);
  const float kn = k * rsqrtf(kss * (1.0f / HD) + RMS_EPS);
  sq[d] = qn;
  sk[d] = kn;
  __syncthreads();

  const size_t ob = ((size_t)h * T_SEQ + t) * HD;
  if (d < 64) {
    // af[j] = (1/1024)^(j/31) for j<32, else 0  -> theta = t * af
    const float theta = (d < 32) ? (float)t * exp2f(-10.0f * (float)d / 31.0f) : 0.0f;
    float s, c;
    __sincosf(theta, &s, &c);
    const float q1 = sq[d], q2 = sq[d + 64];
    Qb[ob + d]      = (__bf16)(q1 * c + q2 * s);
    Qb[ob + d + 64] = (__bf16)(-q1 * s + q2 * c);
    const float k1 = sk[d], k2 = sk[d + 64];
    Kb[ob + d]      = (__bf16)(k1 * c + k2 * s);
    Kb[ob + d + 64] = (__bf16)(-k1 * s + k2 * c);
  }
  const float lam0 = lambdas[0], lam1 = lambdas[1];
  Vb[ob + d] = (__bf16)(lam0 * v + lam1 * ve[(size_t)t * DMODEL + h * HD + d]);
}

// ---------------------------------------------------------------------------
// Kernel 3: flash attention. Block = (head, 128 query rows); 8 waves x 16 rows.
// K chunk staged with CDNA5 async global->LDS copies (ASYNCcnt); V transposed
// through VGPRs (overlaps with in-flight async K copies). WMMA for QK^T / PV.
// ---------------------------------------------------------------------------
__global__ __launch_bounds__(256) void flash_attn(const __bf16* __restrict__ Qb,
                                                  const __bf16* __restrict__ Kb,
                                                  const __bf16* __restrict__ Vb,
                                                  float* __restrict__ Y) {
  __shared__ __bf16 ldsK[32 * 128];   // [key][d]
  __shared__ __bf16 ldsVT[128 * 32];  // [d][key]
  __shared__ __bf16 ldsP[8][16 * 32]; // per-wave P staging (C-layout -> A-layout)

  const int tid  = threadIdx.x;
  const int lane = tid & 31;
  const int w    = tid >> 5;
  const int nl   = lane & 15;
  const int kg   = lane >> 4;
  const int h    = blockIdx.y;
  const int q0   = blockIdx.x * 128;
  const int qrow0 = q0 + w * 16;
  const size_t hb = (size_t)h * T_SEQ * HD;

  // Preload this wave's 16x128 Q tile as four K=32 A-fragments (registers).
  v16bf qf[4];
#pragma unroll
  for (int ds = 0; ds < 4; ++ds)
    qf[ds] = frag_ld(Qb + hb + (size_t)qrow0 * HD, nl, HD, ds * 32, kg);

  v8f yacc[8];
#pragma unroll
  for (int i = 0; i < 8; ++i) yacc[i] = {};
  float mrow[8], lrow[8];
#pragma unroll
  for (int j = 0; j < 8; ++j) { mrow[j] = -1e30f; lrow[j] = 0.0f; }

  const int nchunks = (q0 + 128) >> 5;
  for (int kc = 0; kc < nchunks; ++kc) {
    const int kt0 = kc * 32;
    __syncthreads();
    // stage K chunk [32 x 128] via async global->LDS (no VGPR round-trip)
#pragma unroll
    for (int i = 0; i < 2; ++i) {
      const int e = tid + i * 256;  // 0..511 sixteen-byte packets
      const int key = e >> 4, dg = e & 15;
      const unsigned lds_off =
          (unsigned)(size_t)&ldsK[key * 128 + dg * 8];
      const unsigned long long ga =
          (unsigned long long)(size_t)(Kb + hb + (size_t)(kt0 + key) * HD + dg * 8);
      asm volatile("global_load_async_to_lds_b128 %0, %1, off"
                   :: "v"(lds_off), "v"(ga) : "memory");
    }
    // stage V chunk transposed [128 x 32] (needs VGPR pass for transpose)
#pragma unroll
    for (int i = 0; i < 16; ++i) {
      const int e = tid + i * 256;  // 0..4095
      const int key = e >> 7, d = e & 127;
      ldsVT[d * 32 + key] = Vb[hb + (size_t)(kt0 + key) * HD + d];
    }
    asm volatile("s_wait_asynccnt 0" ::: "memory");
    __syncthreads();

    // scores: S(16x32) = Q(16x128) @ K^T  (two 16-wide N tiles)
    v8f s0 = {}, s1 = {};
#pragma unroll
    for (int ds = 0; ds < 4; ++ds) {
      const v16bf b0 = frag_ld(ldsK, nl, 128, ds * 32, kg);
      const v16bf b1 = frag_ld(ldsK, 16 + nl, 128, ds * 32, kg);
      s0 = wmma_bf16(qf[ds], b0, s0);
      s1 = wmma_bf16(qf[ds], b1, s1);
    }

    float p0[8], p1[8], al[8];
#pragma unroll
    for (int j = 0; j < 8; ++j) {
      const int row = qrow0 + j + 8 * kg;
      float v0 = s0[j] * ATTN_SCALE;
      float v1 = s1[j] * ATTN_SCALE;
      if (kt0 + nl > row)      v0 = -1e30f;  // causal mask
      if (kt0 + 16 + nl > row) v1 = -1e30f;
      // row max across the 16 lanes holding this row (bit4 untouched)
      float mx = fmaxf(v0, v1);
      mx = fmaxf(mx, __shfl_xor(mx, 1));
      mx = fmaxf(mx, __shfl_xor(mx, 2));
      mx = fmaxf(mx, __shfl_xor(mx, 4));
      mx = fmaxf(mx, __shfl_xor(mx, 8));
      const float mnew  = fmaxf(mrow[j], mx);
      const float alpha = __expf(mrow[j] - mnew);
      const float e0 = __expf(v0 - mnew);
      const float e1 = __expf(v1 - mnew);
      float rs = e0 + e1;
      rs += __shfl_xor(rs, 1);
      rs += __shfl_xor(rs, 2);
      rs += __shfl_xor(rs, 4);
      rs += __shfl_xor(rs, 8);
      lrow[j] = lrow[j] * alpha + rs;
      mrow[j] = mnew;
      p0[j] = e0; p1[j] = e1; al[j] = alpha;
    }
    // rescale running PV accumulators
#pragma unroll
    for (int nt = 0; nt < 8; ++nt)
#pragma unroll
      for (int j = 0; j < 8; ++j) yacc[nt][j] *= al[j];

    // P: C-layout -> LDS row-major -> A-fragment
#pragma unroll
    for (int j = 0; j < 8; ++j) {
      ldsP[w][(j + 8 * kg) * 32 + nl]      = (__bf16)p0[j];
      ldsP[w][(j + 8 * kg) * 32 + 16 + nl] = (__bf16)p1[j];
    }
    asm volatile("s_wait_dscnt 0" ::: "memory");  // same-wave LDS RAW
    const v16bf pf = frag_ld(&ldsP[w][0], nl, 32, 0, kg);
#pragma unroll
    for (int nt = 0; nt < 8; ++nt) {
      const v16bf vb = frag_ld(ldsVT, nt * 16 + nl, 32, 0, kg);
      yacc[nt] = wmma_bf16(pf, vb, yacc[nt]);
    }
  }

  // epilogue: divide by softmax denominator, store f32 y[T][768]
#pragma unroll
  for (int j = 0; j < 8; ++j) {
    const float inv = 1.0f / lrow[j];
    const int row = qrow0 + j + 8 * kg;
#pragma unroll
    for (int nt = 0; nt < 8; ++nt)
      Y[(size_t)row * DMODEL + h * HD + nt * 16 + nl] = yacc[nt][j] * inv;
  }
}

// ---------------------------------------------------------------------------
extern "C" void kernel_launch(void* const* d_in, const int* in_sizes, int n_in,
                              void* d_out, int out_size, void* d_ws, size_t ws_size,
                              hipStream_t stream) {
  (void)in_sizes; (void)n_in; (void)out_size; (void)ws_size;
  const float* x       = (const float*)d_in[0];  // [1,4096,768]
  const float* ve      = (const float*)d_in[1];  // [1,4096,768]
  const float* qkv_w   = (const float*)d_in[2];  // [3,768,768] -> [2304,768]
  const float* lambdas = (const float*)d_in[3];  // [2]
  const float* cproj_w = (const float*)d_in[4];  // [768,768]
  float* out = (float*)d_out;                    // [4096,768]

  // workspace layout
  char* ws = (char*)d_ws;
  float* qkv = (float*)ws;                                     // 4096*2304 f32
  const size_t qkv_bytes = (size_t)T_SEQ * 3 * DMODEL * sizeof(float);
  __bf16* Qb = (__bf16*)(ws + qkv_bytes);
  __bf16* Kb = Qb + (size_t)NH * T_SEQ * HD;
  __bf16* Vb = Kb + (size_t)NH * T_SEQ * HD;
  float* y = qkv;  // qkv dead after epilogue; reuse for attention output

  // 1) qkv = x @ qkv_w^T     (M=4096, N=2304, K=768)
  gemm_bf16_wmma<3 * DMODEL, DMODEL>
      <<<dim3((3 * DMODEL) / 64, T_SEQ / 128), 256, 0, stream>>>(x, qkv_w, qkv);

  // 2) rmsnorm + rotary + v-mix -> bf16 Q/K/V head-major
  qkv_epilogue<<<dim3(T_SEQ, NH), 128, 0, stream>>>(qkv, ve, lambdas, Qb, Kb, Vb);

  // 3) flash attention -> y (f32, [4096,768])
  flash_attn<<<dim3(T_SEQ / 128, NH), 256, 0, stream>>>(Qb, Kb, Vb, y);

  // 4) out = y @ c_proj_w^T  (M=4096, N=768, K=768)
  gemm_bf16_wmma<DMODEL, DMODEL>
      <<<dim3(DMODEL / 64, T_SEQ / 128), 256, 0, stream>>>(y, cproj_w, out);
}
